// StDimGlobalLocalContrastModel_62981400428715
// MI455X (gfx1250) — matrix operand order repaired and checked by reading, not compiled, not run on hardware
//
#include <hip/hip_runtime.h>

// ---------------------------------------------------------------------------
// Problem constants (from reference): B=512, LATENT=256, LOCAL=256, HIDDEN=512,
// H*W=256.  logits[s,i,j] = sum_c pred[i,c] * pos[j,c,s], then row-max subtract.
// ---------------------------------------------------------------------------
#define B_N    512
#define LATENT 256
#define LOCAL  256
#define HIDDEN 512
#define HW     256

typedef __attribute__((ext_vector_type(16))) __bf16 v16bf;
typedef __attribute__((ext_vector_type(8)))  float  v8f;
typedef __attribute__((ext_vector_type(4)))  float  v4f;

__device__ __forceinline__ unsigned short f2bf_rne(float x) {
  unsigned u = __float_as_uint(x);
  u += 0x7FFFu + ((u >> 16) & 1u);           // round-to-nearest-even
  return (unsigned short)(u >> 16);
}

// ---------------------------------------------------------------------------
// Tiny MLP chain (0.3 GFLOP total, irrelevant to runtime): scalar f32 kernels.
// ---------------------------------------------------------------------------
__global__ __launch_bounds__(256) void mlp_fc1_relu(
    const float* __restrict__ anchor, const float* __restrict__ W1,
    const float* __restrict__ b1, float* __restrict__ h1) {
  int t = blockIdx.x * 256 + threadIdx.x;     // over 512*512
  int i = t >> 9, n = t & 511;
  const float* ar = anchor + i * LATENT;
  float acc = b1[n];
#pragma unroll 4
  for (int k = 0; k < LATENT; ++k) acc = fmaf(ar[k], W1[k * HIDDEN + n], acc);
  h1[t] = fmaxf(acc, 0.f);
}

__global__ __launch_bounds__(256) void mlp_fc2_res(
    const float* __restrict__ anchor, const float* __restrict__ h1,
    const float* __restrict__ W2, const float* __restrict__ b2,
    float* __restrict__ a) {
  int t = blockIdx.x * 256 + threadIdx.x;     // over 512*256
  int i = t >> 8, n = t & 255;
  float acc = b2[n] + anchor[t];
  const float* hr = h1 + i * HIDDEN;
#pragma unroll 4
  for (int k = 0; k < HIDDEN; ++k) acc = fmaf(hr[k], W2[k * LATENT + n], acc);
  a[t] = acc;
}

// pred = a @ Wc, written in WMMA A-fragment order, bf16:
//   layout [iblk(32)][cc(8)][h(2)][lane(32)][e'(8)]  (16B per (h,lane) chunk)
//   A(16x32 bf16) element: m = lane%16, k = (e%8) + 8*(lane/16) + 16*(e/8)
__global__ __launch_bounds__(256) void pred_fragpack(
    const float* __restrict__ a, const float* __restrict__ Wc,
    unsigned short* __restrict__ predF) {
  int t = blockIdx.x * 256 + threadIdx.x;     // over 512*256
  int m = t >> 8, c = t & 255;
  const float* ar = a + m * LATENT;
  float acc = 0.f;
#pragma unroll 4
  for (int k = 0; k < LATENT; ++k) acc = fmaf(ar[k], Wc[k * LOCAL + c], acc);
  int iblk = m >> 4, n16 = m & 15;
  int cc = c >> 5, crel = c & 31;
  int lanehi = (crel >> 3) & 1;
  int h  = crel >> 4;
  int ep = crel & 7;
  int lane = n16 + (lanehi << 4);
  predF[(((iblk * 8 + cc) * 2 + h) * 32 + lane) * 8 + ep] = f2bf_rne(acc);
}

// ---------------------------------------------------------------------------
// Repack positive[512][256][256] f32 -> bf16 B-fragments (64 MB, one pass).
// B(32x16 bf16) element: n = lane%16, k = (e%8) + 8*(lane/16) + 16*(e/8).
// Output tile layout per (s, jt, cc): [h(2)][lane(32)][e'(8)] bf16 = 1KB,
// tiles at  posB[((s*32+jt)*8+cc) * 1KB]  -> main kernel does 2 coalesced
// b128 loads per tile.  Global reads here are coalesced along s through LDS.
// ---------------------------------------------------------------------------
#define SCH  16   // s-chunk staged in LDS
#define RPAD 20   // padded row stride (floats): 16 data + 4 pad, 16B-aligned
__global__ __launch_bounds__(256) void pos_fragpack(
    const float* __restrict__ pos, unsigned int* __restrict__ posB) {
  __shared__ float brick[512 * RPAD];         // 40 KB
  int jt = blockIdx.x >> 3, cc = blockIdx.x & 7;
  int t  = threadIdx.x;
  int j0 = jt * 16, c0 = cc * 32;
  // write-phase decomposition (constant across s-chunks)
  int h      = t >> 7;                        // 0..1
  int lane   = (t >> 2) & 31;                 // 0..31
  int ep2    = t & 3;                         // bf16-pair index within 16B
  int ep     = ep2 * 2;
  int lanehi = lane >> 4, jrel = lane & 15;
  int crel   = ep + 8 * lanehi + 16 * h;      // k = (e%8) + 8*(lane/16) + 16*(e/8)
  int r_w    = jrel * 32 + crel;

  for (int sc0 = 0; sc0 < HW; sc0 += SCH) {
    // load: 512 (j,c) rows x 16 floats along s, coalesced b128
#pragma unroll
    for (int it = 0; it < 8; ++it) {
      int f = it * 256 + t;
      int row = f >> 2, v4 = f & 3;
      int jr = row >> 5, cr = row & 31;
      const v4f* src = (const v4f*)(pos + (((size_t)(j0 + jr) * LOCAL + (c0 + cr)) * HW + sc0)) + v4;
      *(v4f*)(brick + row * RPAD + v4 * 4) = *src;
    }
    __syncthreads();
    // write: 16 s values, fragment-ordered, coalesced dword stores
    for (int sl = 0; sl < SCH; ++sl) {
      float lo = brick[(r_w)     * RPAD + sl];
      float hi = brick[(r_w + 1) * RPAD + sl];   // crel+1 -> next row
      unsigned int d = (unsigned)f2bf_rne(lo) | ((unsigned)f2bf_rne(hi) << 16);
      int s = sc0 + sl;
      posB[(((size_t)s * 32 + jt) * 8 + cc) * 256 + (h * 32 + lane) * 4 + ep2] = d;
    }
    __syncthreads();
  }
}

// ---------------------------------------------------------------------------
// Main fused kernel: one wave per (s, 16-row i-block).
//  - A fragments (pred, 16x256) resident in 64 VGPRs (L2-hot, 256 KB total)
//  - B tiles double-buffered in registers: next tile's 16 b128 loads issue
//    before the current tile's 8-WMMA chain, so WMMA waits see ~16
//    outstanding loads instead of 0 (full latency hiding)
//  - 16x512 f32 row block parked in 32 KB LDS (CDNA5: 320 KB/WGP)
//  - single LDS read pass: wave32 shfl_xor row-max, subtract, b128 stores
// ---------------------------------------------------------------------------
__device__ __forceinline__ void load_btile(const v4f* __restrict__ tb, int lane,
                                           v16bf b[8]) {
  union AU { v4f f[2]; v16bf v; };
#pragma unroll
  for (int cc = 0; cc < 8; ++cc) {
    AU u;
    u.f[0] = tb[cc * 64 + lane];
    u.f[1] = tb[cc * 64 + 32 + lane];
    b[cc] = u.v;
  }
}

__device__ __forceinline__ void mma_store(const v16bf a[8], const v16bf b[8],
                                          float* Cbuf, int rbase, int col) {
  v8f acc = {0.f, 0.f, 0.f, 0.f, 0.f, 0.f, 0.f, 0.f};
#pragma unroll
  for (int cc = 0; cc < 8; ++cc)
    acc = __builtin_amdgcn_wmma_f32_16x16x32_bf16(
        /*neg_a=*/false, a[cc], /*neg_b=*/false, b[cc],
        /*c_mod=*/(short)0, acc, /*reuse_a=*/false, /*reuse_b=*/false);
#pragma unroll
  for (int v = 0; v < 8; ++v) Cbuf[(rbase + v) * 512 + col] = acc[v];
}

__global__ __launch_bounds__(32) void gemm_rowmax(
    const unsigned short* __restrict__ predF,
    const v4f* __restrict__ posB4,
    float* __restrict__ out) {
  int iblk = blockIdx.x & 31;                 // i-blocks fastest: 32 consecutive
  int s    = blockIdx.x >> 5;                 // blocks reuse one 256KB posB slice
  int lane = threadIdx.x;

  union AU { v4f f[2]; v16bf v; };

  // preload all 8 A fragments for this i-block
  v16bf afrag[8];
  const v4f* pA = (const v4f*)predF;
#pragma unroll
  for (int cc = 0; cc < 8; ++cc) {
    AU u;
    u.f[0] = pA[((iblk * 8 + cc) * 2 + 0) * 32 + lane];
    u.f[1] = pA[((iblk * 8 + cc) * 2 + 1) * 32 + lane];
    afrag[cc] = u.v;
  }

  __shared__ float Cbuf[16 * 512];            // 32 KB row buffer
  const v4f* sbase = posB4 + (size_t)s * (32 * 8 * 64);
  int rbase = (lane >> 4) * 8;                // C layout: lanes 16-31 hold M+8
  int nlo   = lane & 15;

  v16bf b0[8], b1[8];
  load_btile(sbase, lane, b0);                // tile 0 in flight
  for (int jt = 0; jt < 32; jt += 2) {
    if (jt + 4 < 32)
      __builtin_prefetch((const void*)(sbase + (jt + 4) * 512), 0, 1);
    load_btile(sbase + (jt + 1) * 512, lane, b1);     // next tile in flight
    mma_store(afrag, b0, Cbuf, rbase, jt * 16 + nlo); // compute current
    if (jt + 2 < 32)
      load_btile(sbase + (jt + 2) * 512, lane, b0);   // tile after next
    mma_store(afrag, b1, Cbuf, rbase, (jt + 1) * 16 + nlo);
  }

  // row max over j (512), subtract, store (coalesced b128)
  const v4f* C4 = (const v4f*)Cbuf;
  size_t obase = ((size_t)s * B_N + (size_t)iblk * 16) * B_N;
  for (int r = 0; r < 16; ++r) {
    v4f x0 = C4[r * 128 +      lane];
    v4f x1 = C4[r * 128 + 32 + lane];
    v4f x2 = C4[r * 128 + 64 + lane];
    v4f x3 = C4[r * 128 + 96 + lane];
    float m = fmaxf(fmaxf(fmaxf(x0.x, x0.y), fmaxf(x0.z, x0.w)),
                    fmaxf(fmaxf(x1.x, x1.y), fmaxf(x1.z, x1.w)));
    m = fmaxf(m, fmaxf(fmaxf(fmaxf(x2.x, x2.y), fmaxf(x2.z, x2.w)),
                       fmaxf(fmaxf(x3.x, x3.y), fmaxf(x3.z, x3.w))));
#pragma unroll
    for (int off = 16; off >= 1; off >>= 1)
      m = fmaxf(m, __shfl_xor(m, off, 32));   // wave32 reduction
    v4f* orow = (v4f*)(out + obase + (size_t)r * B_N);
    x0 -= m; x1 -= m; x2 -= m; x3 -= m;
    orow[     lane] = x0;
    orow[32 + lane] = x1;
    orow[64 + lane] = x2;
    orow[96 + lane] = x3;
  }
}

// ---------------------------------------------------------------------------
// Launch.  ws layout: [0,64MB) posB bf16 fragments | +256KB predF | +1MB h1
//          | +512KB a   (total ~65.8 MB of scratch)
// ---------------------------------------------------------------------------
extern "C" void kernel_launch(void* const* d_in, const int* in_sizes, int n_in,
                              void* d_out, int out_size, void* d_ws, size_t ws_size,
                              hipStream_t stream) {
  const float* anchor = (const float*)d_in[0];
  const float* pos    = (const float*)d_in[1];
  const float* W1     = (const float*)d_in[2];
  const float* b1     = (const float*)d_in[3];
  const float* W2     = (const float*)d_in[4];
  const float* b2     = (const float*)d_in[5];
  const float* Wc     = (const float*)d_in[6];

  char* ws = (char*)d_ws;
  unsigned int*   posB  = (unsigned int*)ws;                                   // 64 MiB
  unsigned short* predF = (unsigned short*)(ws + (64ull << 20));               // 256 KiB
  float*          h1    = (float*)(ws + (64ull << 20) + (256ull << 10));       // 1 MiB
  float*          a     = (float*)(ws + (64ull << 20) + (256ull << 10) + (1ull << 20)); // 512 KiB

  mlp_fc1_relu <<<(B_N * HIDDEN) / 256, 256, 0, stream>>>(anchor, W1, b1, h1);
  mlp_fc2_res  <<<(B_N * LATENT) / 256, 256, 0, stream>>>(anchor, h1, W2, b2, a);
  pred_fragpack<<<(B_N * LOCAL)  / 256, 256, 0, stream>>>(a, Wc, predF);
  pos_fragpack <<<32 * 8, 256, 0, stream>>>(pos, posB);
  gemm_rowmax  <<<HW * (B_N / 16), 32, 0, stream>>>(predF, (const v4f*)posB,
                                                    (float*)d_out);
}
